// PositionAwareAttentionModule_2276332667081
// MI455X (gfx1250) — matrix operand order repaired and checked by reading, not compiled, not run on hardware
//
#include <hip/hip_runtime.h>

// ---------------------------------------------------------------------------
// Problem constants
// ---------------------------------------------------------------------------
#define BATCH 4
#define C_IN 256
#define IC 128
#define NSPAT 4096          // H*W = 64*64
#define EPS 1e-5f

typedef __attribute__((ext_vector_type(16))) __bf16 v16bf;
typedef __attribute__((ext_vector_type(8)))  __bf16 v8bf;
typedef __attribute__((ext_vector_type(8)))  float  v8f;
typedef __attribute__((ext_vector_type(4)))  unsigned int u32x4;
typedef __attribute__((ext_vector_type(8)))  unsigned int u32x8;

// ---------------------------------------------------------------------------
// Fragment helpers (CDNA5 WMMA 16x16x32 bf16, wave32)
//
// A-matrix (16xK, row-major source, per 32-K slice):
//   lane l: row = l%16 ; elements 0..7  = K kb..kb+7
//                         elements 8..15 = K kb+16..kb+23,  kb = 8*(l/16)
// B-matrix (Kx16, source laid out so K is contiguous per column):
//   lane l: col = l%16 ; same K mapping as A (kb..kb+7, kb+16..kb+23), kb = 8*(l/16)
// C/D: vgpr r -> row r (lanes 0..15) / row r+8 (lanes 16..31); col = l%16
// ---------------------------------------------------------------------------
__device__ __forceinline__ v16bf cat8(v8bf lo, v8bf hi) {
  return __builtin_shufflevector(lo, hi, 0,1,2,3,4,5,6,7,8,9,10,11,12,13,14,15);
}
__device__ __forceinline__ v16bf load_fragA(const __bf16* p) {
  v8bf lo = *(const v8bf*)p;
  v8bf hi = *(const v8bf*)(p + 16);
  return cat8(lo, hi);
}
__device__ __forceinline__ v16bf load_fragB(const __bf16* p) {
  v8bf lo = *(const v8bf*)p;
  v8bf hi = *(const v8bf*)(p + 16);
  return cat8(lo, hi);
}
__device__ __forceinline__ v8f wmma_bf16(v16bf a, v16bf b, v8f c) {
  return __builtin_amdgcn_wmma_f32_16x16x32_bf16(false, a, false, b,
                                                 (short)0, c, false, false);
}

// ---------------------------------------------------------------------------
// Kernel 0: fold BN into effective weights (bf16) and biases (f32)
// ---------------------------------------------------------------------------
__global__ void prep_weights(const float* __restrict__ w, const float* __restrict__ b,
                             const float* __restrict__ gamma, const float* __restrict__ beta,
                             const float* __restrict__ mean, const float* __restrict__ var,
                             __bf16* __restrict__ weff, float* __restrict__ beff,
                             int co, int ci) {
  int i = blockIdx.x * 256 + threadIdx.x;
  if (i < co * ci) {
    int o = i / ci;
    float s = gamma[o] * rsqrtf(var[o] + EPS);
    weff[i] = (__bf16)(w[i] * s);
  }
  if (i < co) {
    float s = gamma[i] * rsqrtf(var[i] + EPS);
    beff[i] = (b[i] - mean[i]) * s + beta[i];
  }
}

// ---------------------------------------------------------------------------
// Kernel 1: projections g/theta/phi = relu(Weff * x + beff)
//   g stored [c][n]   (TDM-friendly + B frags along m in attention)
//   theta, phi stored TRANSPOSED [n][c] so the attention kernel can load
//   A/B fragments straight from global memory with zero re-layout.
// ---------------------------------------------------------------------------
#define XT_STRIDE 272
__global__ void proj_kernel(const float* __restrict__ x,
                            const __bf16* __restrict__ wg, const __bf16* __restrict__ wth,
                            const __bf16* __restrict__ wph,
                            const float* __restrict__ bg, const float* __restrict__ bth,
                            const float* __restrict__ bph,
                            __bf16* __restrict__ og, __bf16* __restrict__ oth,
                            __bf16* __restrict__ oph) {
  __shared__ __bf16 xT[64 * XT_STRIDE];

  const int n0 = blockIdx.x * 64;
  const int b  = blockIdx.y;
  const float* xb = x + (size_t)b * C_IN * NSPAT;

  {
    int n = threadIdx.x & 63;
    int cb = threadIdx.x >> 6;            // 0..3
#pragma unroll 4
    for (int k = 0; k < 64; ++k) {
      int c = cb + 4 * k;
      xT[n * XT_STRIDE + c] = (__bf16)xb[(size_t)c * NSPAT + n0 + n];
    }
  }
  __syncthreads();

  const int lane = threadIdx.x & 31;
  const int wave = threadIdx.x >> 5;
  const int lh   = lane >> 4;
  const int ll   = lane & 15;

  const __bf16* W[3] = { wg, wth, wph };
  const float*  Bv[3] = { bg, bth, bph };
  __bf16* O[3];
  O[0] = og  + (size_t)b * IC * NSPAT;    // [c][n]
  O[1] = oth + (size_t)b * NSPAT * IC;    // [n][c]
  O[2] = oph + (size_t)b * NSPAT * IC;    // [n][c]

  const int row = wave * 16 + ll;

#pragma unroll
  for (int br = 0; br < 3; ++br) {
    v8f acc[4];
#pragma unroll
    for (int f = 0; f < 4; ++f) acc[f] = (v8f){0,0,0,0,0,0,0,0};

#pragma unroll
    for (int k = 0; k < 8; ++k) {
      v16bf a = load_fragA(W[br] + (size_t)row * C_IN + k * 32 + 8 * lh);
#pragma unroll
      for (int f = 0; f < 4; ++f) {
        v16bf bfrag = load_fragB(&xT[(f * 16 + ll) * XT_STRIDE + k * 32 + 8 * lh]);
        acc[f] = wmma_bf16(a, bfrag, acc[f]);
      }
    }
#pragma unroll
    for (int f = 0; f < 4; ++f) {
#pragma unroll
      for (int r = 0; r < 8; ++r) {
        int o = wave * 16 + r + 8 * lh;   // output channel
        int n = n0 + f * 16 + ll;         // spatial position
        float v = acc[f][r] + Bv[br][o];
        v = fmaxf(v, 0.0f);
        if (br == 0)
          O[0][(size_t)o * NSPAT + n] = (__bf16)v;          // g: [c][n]
        else
          O[br][(size_t)n * IC + o] = (__bf16)v;            // theta/phi: [n][c]
      }
    }
  }
}

// ---------------------------------------------------------------------------
// Kernel 2: fused flash attention.
//   S = thetaT * phi (K=128), online softmax over m, acc += P * g^T
// thetaT/phiT are [n][c]: A and B fragments come straight from global.
// g tile [128 c][32 m] is DMA'd to LDS by the Tensor Data Mover each step.
// LDS use: 8KB g tile + 8KB per-wave P staging -> high WGP occupancy.
// ---------------------------------------------------------------------------
__global__ void attn_kernel(const __bf16* __restrict__ thetaT,
                            const __bf16* __restrict__ phiT,
                            const __bf16* __restrict__ g,
                            __bf16* __restrict__ y) {
  __shared__ __bf16 ldsG[128 * 32];               // [c][m]
  __shared__ __bf16 ldsP[8 * 16 * 32];            // per-wave P staging [16][32]

  const int n0 = blockIdx.x * 128;
  const int b  = blockIdx.y;
  const __bf16* th_b  = thetaT + (size_t)b * NSPAT * IC;
  const __bf16* phi_b = phiT   + (size_t)b * NSPAT * IC;
  const __bf16* g_b   = g      + (size_t)b * IC * NSPAT;
  __bf16* y_b = y + (size_t)b * NSPAT * IC;

  const int lane = threadIdx.x & 31;
  const int wave = threadIdx.x >> 5;
  const int lh = lane >> 4;
  const int ll = lane & 15;

  // per-wave A fragments of theta^T (K = 128 -> 4 slices): direct from global
  v16bf Ath[4];
#pragma unroll
  for (int k = 0; k < 4; ++k)
    Ath[k] = load_fragA(th_b + (size_t)(n0 + wave * 16 + ll) * IC + k * 32 + 8 * lh);

  v8f acc[8];
#pragma unroll
  for (int t = 0; t < 8; ++t) acc[t] = (v8f){0,0,0,0,0,0,0,0};
  float rowmax[8], rowsum[8];
#pragma unroll
  for (int r = 0; r < 8; ++r) { rowmax[r] = -1.0e30f; rowsum[r] = 0.0f; }

  __bf16* Pw = &ldsP[wave * 16 * 32];
  const unsigned int ldsG_addr = (unsigned int)(uintptr_t)(&ldsG[0]);

  for (int mi = 0; mi < NSPAT / 32; ++mi) {
    const int m0 = mi * 32;
    __syncthreads();                      // prior readers of ldsG done

    // --- TDM: DMA the g tile [128 c][32 m] into LDS (wave 0 issues) -------
    if (wave == 0) {
      unsigned long long gaddr = (unsigned long long)(uintptr_t)(g_b + m0);
      u32x4 d0;
      d0[0] = 1u;                                   // count=1 (valid D#)
      d0[1] = ldsG_addr;                            // lds_addr (bytes)
      d0[2] = (unsigned int)gaddr;                  // global_addr[31:0]
      d0[3] = (unsigned int)(gaddr >> 32) | (2u << 30);  // addr[56:32] | type=2
      u32x8 d1;
      d1[0] = 1u << 16;                             // data_size = 2 bytes
      d1[1] = (NSPAT & 0xffffu) << 16;              // tensor_dim0 lo16 (4096)
      d1[2] = (NSPAT >> 16) | ((IC & 0xffffu) << 16); // dim0 hi16 | tensor_dim1 lo16
      d1[3] = (32u << 16);                          // dim1 hi16 | tile_dim0 = 32
      d1[4] = 128u;                                 // tile_dim1 = 128, tile_dim2 = 0
      d1[5] = (unsigned int)NSPAT;                  // tensor_dim0_stride lo32
      d1[6] = 0u;
      d1[7] = 0u;
      asm volatile("tensor_load_to_lds %0, %1" :: "s"(d0), "s"(d1) : "memory");
      __builtin_amdgcn_s_wait_tensorcnt(0);
    }
    if (mi + 1 < NSPAT / 32)
      __builtin_prefetch(phi_b + (size_t)(m0 + 32) * IC, 0, 1);
    __syncthreads();                      // ldsG tile visible to all waves

    // S tile 16(n) x 32(m); B fragments straight from global phiT [m][c]
    v8f S0 = (v8f){0,0,0,0,0,0,0,0};
    v8f S1 = (v8f){0,0,0,0,0,0,0,0};
#pragma unroll
    for (int k = 0; k < 4; ++k) {
      v16bf b0 = load_fragB(phi_b + (size_t)(m0 + ll) * IC + k * 32 + 8 * lh);
      S0 = wmma_bf16(Ath[k], b0, S0);
      v16bf b1 = load_fragB(phi_b + (size_t)(m0 + 16 + ll) * IC + k * 32 + 8 * lh);
      S1 = wmma_bf16(Ath[k], b1, S1);
    }

    // online softmax per row (rows live across 16-lane halves)
    float scale[8];
#pragma unroll
    for (int r = 0; r < 8; ++r) {
      float v = fmaxf(S0[r], S1[r]);
      v = fmaxf(v, __shfl_xor(v, 1, 32));
      v = fmaxf(v, __shfl_xor(v, 2, 32));
      v = fmaxf(v, __shfl_xor(v, 4, 32));
      v = fmaxf(v, __shfl_xor(v, 8, 32));
      float nm = fmaxf(rowmax[r], v);
      float corr = __expf(rowmax[r] - nm);
      rowmax[r] = nm;
      float e0 = __expf(S0[r] - nm);
      float e1 = __expf(S1[r] - nm);
      float s = e0 + e1;
      s += __shfl_xor(s, 1, 32);
      s += __shfl_xor(s, 2, 32);
      s += __shfl_xor(s, 4, 32);
      s += __shfl_xor(s, 8, 32);
      rowsum[r] = rowsum[r] * corr + s;
      scale[r] = corr;
      int prow = r + 8 * lh;
      Pw[prow * 32 + ll]      = (__bf16)e0;
      Pw[prow * 32 + 16 + ll] = (__bf16)e1;
    }
#pragma unroll
    for (int t = 0; t < 8; ++t)
#pragma unroll
      for (int r = 0; r < 8; ++r) acc[t][r] *= scale[r];

    // wave-private LDS relayout fence (C layout -> A layout)
    asm volatile("s_wait_dscnt 0" ::: "memory");
    v16bf Pfrag = load_fragA(&Pw[ll * 32 + 8 * lh]);

#pragma unroll
    for (int t = 0; t < 8; ++t) {
      v16bf bg = load_fragB(&ldsG[(t * 16 + ll) * 32 + 8 * lh]);
      acc[t] = wmma_bf16(Pfrag, bg, acc[t]);
    }
  }

  // normalize and store y as bf16 [n][c]
  float inv[8];
#pragma unroll
  for (int r = 0; r < 8; ++r) inv[r] = 1.0f / rowsum[r];
#pragma unroll
  for (int t = 0; t < 8; ++t) {
#pragma unroll
    for (int r = 0; r < 8; ++r) {
      int n = n0 + wave * 16 + r + 8 * lh;
      int c = t * 16 + ll;
      y_b[(size_t)n * IC + c] = (__bf16)(acc[t][r] * inv[r]);
    }
  }
}

// ---------------------------------------------------------------------------
// Kernel 3: out = Wf_eff * y + bf_eff + x   (f32 output, residual add)
// ---------------------------------------------------------------------------
__global__ void outconv_kernel(const __bf16* __restrict__ y,
                               const __bf16* __restrict__ wf,
                               const float* __restrict__ bf,
                               const float* __restrict__ x,
                               float* __restrict__ out) {
  const int n0  = blockIdx.x * 64;
  const int co0 = blockIdx.y * 128;
  const int b   = blockIdx.z;
  const __bf16* y_b = y + (size_t)b * NSPAT * IC;

  const int lane = threadIdx.x & 31;
  const int wave = threadIdx.x >> 5;
  const int lh = lane >> 4;
  const int ll = lane & 15;
  const int row = co0 + wave * 16 + ll;

  v8f acc[4];
#pragma unroll
  for (int f = 0; f < 4; ++f) acc[f] = (v8f){0,0,0,0,0,0,0,0};

#pragma unroll
  for (int k = 0; k < 4; ++k) {
    v16bf a = load_fragA(wf + (size_t)row * IC + k * 32 + 8 * lh);
#pragma unroll
    for (int f = 0; f < 4; ++f) {
      v16bf bfrag = load_fragB(y_b + (size_t)(n0 + f * 16 + ll) * IC + k * 32 + 8 * lh);
      acc[f] = wmma_bf16(a, bfrag, acc[f]);
    }
  }
#pragma unroll
  for (int f = 0; f < 4; ++f) {
#pragma unroll
    for (int r = 0; r < 8; ++r) {
      int co = co0 + wave * 16 + r + 8 * lh;
      int n  = n0 + f * 16 + ll;
      size_t idx = ((size_t)b * C_IN + co) * NSPAT + n;
      out[idx] = acc[f][r] + bf[co] + x[idx];
    }
  }
}

// ---------------------------------------------------------------------------
// Host-side orchestration
// ---------------------------------------------------------------------------
extern "C" void kernel_launch(void* const* d_in, const int* in_sizes, int n_in,
                              void* d_out, int out_size, void* d_ws, size_t ws_size,
                              hipStream_t stream) {
  const float* x = (const float*)d_in[0];
  const float* g_w  = (const float*)d_in[1];
  const float* th_w = (const float*)d_in[7];
  const float* ph_w = (const float*)d_in[13];
  const float* w_w  = (const float*)d_in[19];

  char* ws = (char*)d_ws;
  __bf16* WG  = (__bf16*)(ws + 0);
  __bf16* WTH = (__bf16*)(ws + 65536);
  __bf16* WPH = (__bf16*)(ws + 131072);
  __bf16* WF  = (__bf16*)(ws + 196608);
  float*  BG  = (float*)(ws + 262144);
  float*  BTH = (float*)(ws + 263168);
  float*  BPH = (float*)(ws + 264192);
  float*  BF  = (float*)(ws + 265216);
  __bf16* THETA = (__bf16*)(ws + 524288);               // [B][n][c] transposed
  __bf16* PHI   = (__bf16*)(ws + 524288 + 4194304);     // [B][n][c] transposed
  __bf16* GX    = (__bf16*)(ws + 524288 + 2 * 4194304); // [B][c][n]
  __bf16* Y     = (__bf16*)(ws + 524288 + 3 * 4194304); // [B][n][c]

  prep_weights<<<128, 256, 0, stream>>>(g_w,  (const float*)d_in[2],
      (const float*)d_in[3], (const float*)d_in[4], (const float*)d_in[5],
      (const float*)d_in[6], WG, BG, IC, C_IN);
  prep_weights<<<128, 256, 0, stream>>>(th_w, (const float*)d_in[8],
      (const float*)d_in[9], (const float*)d_in[10], (const float*)d_in[11],
      (const float*)d_in[12], WTH, BTH, IC, C_IN);
  prep_weights<<<128, 256, 0, stream>>>(ph_w, (const float*)d_in[14],
      (const float*)d_in[15], (const float*)d_in[16], (const float*)d_in[17],
      (const float*)d_in[18], WPH, BPH, IC, C_IN);
  prep_weights<<<128, 256, 0, stream>>>(w_w,  (const float*)d_in[20],
      (const float*)d_in[21], (const float*)d_in[22], (const float*)d_in[23],
      (const float*)d_in[24], WF, BF, C_IN, IC);

  proj_kernel<<<dim3(NSPAT / 64, BATCH), 256, 0, stream>>>(
      x, WG, WTH, WPH, BG, BTH, BPH, GX, THETA, PHI);

  attn_kernel<<<dim3(NSPAT / 128, BATCH), 256, 0, stream>>>(THETA, PHI, GX, Y);

  outconv_kernel<<<dim3(NSPAT / 64, C_IN / 128, BATCH), 256, 0, stream>>>(
      Y, WF, BF, x, (float*)d_out);
}
